// SwinTransformerBlock_7962869367122
// MI455X (gfx1250) — matrix operand order repaired and checked by reading, not compiled, not run on hardware
//
#include <hip/hip_runtime.h>
#include <hip/hip_bf16.h>
#include <math.h>

// ---------- types ----------
typedef __attribute__((ext_vector_type(16))) _Float16 v16h;
typedef __attribute__((ext_vector_type(8)))  _Float16 v8h;
typedef __attribute__((ext_vector_type(8)))  float    v8f;

#define WMMA_F16(a, b, c) \
  __builtin_amdgcn_wmma_f32_16x16x32_f16(false, (a), false, (b), (short)0, (c), false, false)

__device__ __forceinline__ v16h cat8(v8h lo, v8h hi) {
  return __builtin_shufflevector(lo, hi, 0,1,2,3,4,5,6,7,8,9,10,11,12,13,14,15);
}

// ---------- problem constants (H = W = 56 per setup_inputs) ----------
#define C_DIM   384
#define NH      12
#define HD      32
#define WS7     7
#define NTOK    49        // tokens per window
#define NPAD    64        // padded rows per window
#define SS3     3
#define HW      56
#define NWIN    64        // 8x8 windows per image
#define BATCH   32
#define NWINTOT (BATCH * NWIN)            // 2048
#define MPAD    ((size_t)NWINTOT * NPAD)  // 131072 padded GEMM rows
#define NTOKENS (BATCH * HW * HW)         // 100352
#define SCALE_Q 0.17677669529663687f      // (C/NH)^-0.5 = 32^-0.5

// =====================================================================
// util kernels
// =====================================================================
__global__ void fill_zero4(uint4* p, size_t n4) {
  size_t i = (size_t)blockIdx.x * blockDim.x + threadIdx.x;
  if (i < n4) p[i] = make_uint4(0u, 0u, 0u, 0u);
}

__global__ void cvt_f32_f16(const float* __restrict__ src, _Float16* __restrict__ dst, int n) {
  int i = blockIdx.x * blockDim.x + threadIdx.x;
  if (i < n) dst[i] = (_Float16)src[i];
}

// =====================================================================
// LN1 + shift + window partition.  One wave per token, 8 waves / block.
// dest row m = (b*64 + win)*64 + n  (n = window-local token)
// =====================================================================
__global__ __launch_bounds__(256) void ln1_shift_window_kernel(
    const float* __restrict__ x, const float* __restrict__ g, const float* __restrict__ b,
    _Float16* __restrict__ xw) {
  const int token = blockIdx.x * 8 + (threadIdx.x >> 5);
  const int lane  = threadIdx.x & 31;
  const float* xr = x + (size_t)token * C_DIM;
  float v[12], s = 0.f, s2 = 0.f;
#pragma unroll
  for (int i = 0; i < 12; ++i) { v[i] = xr[lane * 12 + i]; s += v[i]; s2 += v[i] * v[i]; }
#pragma unroll
  for (int o = 16; o > 0; o >>= 1) { s += __shfl_xor(s, o, 32); s2 += __shfl_xor(s2, o, 32); }
  const float mu = s * (1.f / C_DIM);
  const float rs = rsqrtf(s2 * (1.f / C_DIM) - mu * mu + 1e-5f);

  const int bi = token / (HW * HW);
  const int hw = token % (HW * HW);
  int hp = hw / HW - SS3; if (hp < 0) hp += HW;
  int wp = hw % HW - SS3; if (wp < 0) wp += HW;
  const int win = (hp / WS7) * 8 + (wp / WS7);
  const int n   = (hp % WS7) * WS7 + (wp % WS7);
  const size_t m = ((size_t)bi * NWIN + win) * NPAD + n;
  _Float16* dst = xw + m * C_DIM + lane * 12;
#pragma unroll
  for (int i = 0; i < 12; ++i) {
    const int c = lane * 12 + i;
    dst[i] = (_Float16)((v[i] - mu) * rs * g[c] + b[c]);
  }
}

// =====================================================================
// WMMA GEMM core: one wave computes a 32(M) x 64(N) tile of
//   C = A[M,K] * W^T   (W row-major [N,K] f16 -> native B-fragment layout)
// 8 WMMAs per 32-wide K-step against 6 fragment loads (2 A + 4 B).
// =====================================================================
__device__ __forceinline__ void wave_gemm_32x64(
    const _Float16* __restrict__ A, int lda,
    const _Float16* __restrict__ W, int ldw,
    int K, size_t mBase, int nBase, int lane, v8f acc[2][4]) {
  const int lrow = lane & 15;
  const int lhi  = lane >> 4;
  const _Float16* arow0 = A + (mBase + lrow) * (size_t)lda;
  const _Float16* arow1 = A + (mBase + 16 + lrow) * (size_t)lda;
  for (int k0 = 0; k0 < K; k0 += 32) {
    // A 16x32 fragment: elem i -> K = k0 + lhi*16 + i
    v16h a0 = cat8(*(const v8h*)(arow0 + k0 + lhi * 16),
                   *(const v8h*)(arow0 + k0 + lhi * 16 + 8));
    v16h a1 = cat8(*(const v8h*)(arow1 + k0 + lhi * 16),
                   *(const v8h*)(arow1 + k0 + lhi * 16 + 8));
    if (k0 + 32 < K) {
      __builtin_prefetch(arow0 + k0 + 32, 0, 1); // global_prefetch_b8
      __builtin_prefetch(arow1 + k0 + 32, 0, 1);
    }
#pragma unroll
    for (int j = 0; j < 4; ++j) {
      // B 32x16 fragment: lane col n, 16 contiguous K halfs at k0 + lhi*16
      const _Float16* bp = W + (size_t)(nBase + j * 16 + lrow) * ldw + k0 + lhi * 16;
      v16h bf = *(const v16h*)bp;
      acc[0][j] = WMMA_F16(a0, bf, acc[0][j]);
      acc[1][j] = WMMA_F16(a1, bf, acc[1][j]);
    }
  }
}

#define GEMM_PROLOG(LDA)                                              \
  const int lane = threadIdx.x & 31, wave = threadIdx.x >> 5;         \
  const size_t mBase = (size_t)blockIdx.y * 128 + (wave >> 1) * 32;   \
  const int    nBase = blockIdx.x * 128 + (wave & 1) * 64;            \
  v8f z = {};                                                         \
  v8f acc[2][4] = {{z, z, z, z}, {z, z, z, z}};                       \
  const int lrow = lane & 15, lhi = lane >> 4;                        \
  (void)lrow; (void)lhi;

// ---------------- QKV GEMM: A[131072,384] x qkv_w^T -> q/k/vT (f16) --------
__global__ __launch_bounds__(256) void gemm_qkv_kernel(
    const _Float16* __restrict__ A, const _Float16* __restrict__ W,
    const float* __restrict__ bias,
    _Float16* __restrict__ qb, _Float16* __restrict__ kb, _Float16* __restrict__ vT) {
  GEMM_PROLOG(C_DIM)
  wave_gemm_32x64(A, C_DIM, W, C_DIM, C_DIM, mBase, nBase, lane, acc);
#pragma unroll
  for (int j = 0; j < 4; ++j) {
    const int c    = nBase + j * 16 + lrow;     // 0..1151
    const int sel  = c / C_DIM;                 // 0=q 1=k 2=v
    const int head = (c % C_DIM) >> 5;
    const int d    = c & 31;
    const float bv = bias[c];
    const float sc = (sel == 0) ? SCALE_Q : 1.0f;
#pragma unroll
    for (int t = 0; t < 2; ++t) {
#pragma unroll
      for (int r = 0; r < 8; ++r) {
        const size_t Mg  = mBase + t * 16 + r + lhi * 8;
        const size_t win = Mg >> 6;
        const int    n   = (int)(Mg & 63);
        const float  val = (acc[t][j][r] + bv) * sc;
        if (sel == 0)      qb[(((win * NH) + head) * NPAD + n) * HD + d] = (_Float16)val;
        else if (sel == 1) kb[(((win * NH) + head) * NPAD + n) * HD + d] = (_Float16)val;
        else               vT[(((win * NH) + head) * HD + d) * NPAD + n] = (_Float16)val;
      }
    }
  }
}

// ---------------- proj GEMM -> fp32 ---------------------------------------
__global__ __launch_bounds__(256) void gemm_proj_kernel(
    const _Float16* __restrict__ A, const _Float16* __restrict__ W,
    const float* __restrict__ bias, float* __restrict__ out) {
  GEMM_PROLOG(C_DIM)
  wave_gemm_32x64(A, C_DIM, W, C_DIM, C_DIM, mBase, nBase, lane, acc);
#pragma unroll
  for (int j = 0; j < 4; ++j) {
    const int c = nBase + j * 16 + lrow;
    const float bv = bias[c];
#pragma unroll
    for (int t = 0; t < 2; ++t)
#pragma unroll
      for (int r = 0; r < 8; ++r) {
        const size_t Mg = mBase + t * 16 + r + lhi * 8;
        out[Mg * C_DIM + c] = acc[t][j][r] + bv;
      }
  }
}

// ---------------- fc1 GEMM + exact GELU -----------------------------------
__global__ __launch_bounds__(256) void gemm_fc1_gelu_kernel(
    const _Float16* __restrict__ A, const _Float16* __restrict__ W,
    const float* __restrict__ bias, _Float16* __restrict__ out) {
  GEMM_PROLOG(C_DIM)
  wave_gemm_32x64(A, C_DIM, W, C_DIM, C_DIM, mBase, nBase, lane, acc);
#pragma unroll
  for (int j = 0; j < 4; ++j) {
    const int c = nBase + j * 16 + lrow;
    const float bv = bias[c];
#pragma unroll
    for (int t = 0; t < 2; ++t)
#pragma unroll
      for (int r = 0; r < 8; ++r) {
        const size_t Mg = mBase + t * 16 + r + lhi * 8;
        const float v = acc[t][j][r] + bv;
        out[Mg * (4 * C_DIM) + c] = (_Float16)(v * 0.5f * (1.f + erff(v * 0.70710678118f)));
      }
  }
}

// ---------------- fc2 GEMM + residual -> final output ----------------------
__global__ __launch_bounds__(256) void gemm_fc2_res_kernel(
    const _Float16* __restrict__ A, const _Float16* __restrict__ W,
    const float* __restrict__ bias, const float* __restrict__ x1,
    float* __restrict__ out) {
  GEMM_PROLOG(4 * C_DIM)
  wave_gemm_32x64(A, 4 * C_DIM, W, 4 * C_DIM, 4 * C_DIM, mBase, nBase, lane, acc);
#pragma unroll
  for (int j = 0; j < 4; ++j) {
    const int c = nBase + j * 16 + lrow;
    const float bv = bias[c];
#pragma unroll
    for (int t = 0; t < 2; ++t)
#pragma unroll
      for (int r = 0; r < 8; ++r) {
        const size_t Mg = mBase + t * 16 + r + lhi * 8;
        out[Mg * C_DIM + c] = x1[Mg * C_DIM + c] + acc[t][j][r] + bv;
      }
  }
}

// =====================================================================
// windowed attention: one wave per (window, head)
// =====================================================================
__device__ __forceinline__ int regid(int t) { return t < 49 ? 0 : (t < 53 ? 1 : 2); }

__global__ __launch_bounds__(32) void attn_kernel(
    const _Float16* __restrict__ q, const _Float16* __restrict__ k,
    const _Float16* __restrict__ vT, const float* __restrict__ rpb,
    _Float16* __restrict__ ao) {
  __shared__ __align__(16) float    S[NPAD * NPAD];
  __shared__ __align__(16) _Float16 P[NPAD * NPAD];

  const int wh   = blockIdx.x;          // 0..24575
  const int win  = wh / NH;
  const int head = wh % NH;
  const int lane = threadIdx.x;
  const int lrow = lane & 15, lhi = lane >> 4;

  const _Float16* qbase = q  + ((size_t)(win * NH + head)) * NPAD * HD;
  const _Float16* kbase = k  + ((size_t)(win * NH + head)) * NPAD * HD;
  const _Float16* vbase = vT + ((size_t)(win * NH + head)) * HD * NPAD;

  // A-fragments of q (K = hd = 32: one WMMA K-step)
  v16h qf[4];
#pragma unroll
  for (int mt = 0; mt < 4; ++mt) {
    const _Float16* p = qbase + (mt * 16 + lrow) * HD + lhi * 16;
    qf[mt] = cat8(*(const v8h*)p, *(const v8h*)(p + 8));
  }
  // B-fragments of k^T: b[i] = K[m = nt*16+lrow][d = lhi*16 + i]
  v16h kf[4];
#pragma unroll
  for (int nt = 0; nt < 4; ++nt)
    kf[nt] = *(const v16h*)(kbase + (nt * 16 + lrow) * HD + lhi * 16);

  const int widx = win & 63;
  const int wwh = widx >> 3, www = widx & 7;

  // S = q k^T + bias + mask  -> LDS (fp32)
#pragma unroll
  for (int mt = 0; mt < 4; ++mt) {
#pragma unroll
    for (int nt = 0; nt < 4; ++nt) {
      v8f acc = {};
      acc = WMMA_F16(qf[mt], kf[nt], acc);
#pragma unroll
      for (int r = 0; r < 8; ++r) {
        const int nq = mt * 16 + r + lhi * 8;   // query token
        const int mk = nt * 16 + lrow;          // key token
        float sval = -1e30f;
        if (nq < NTOK && mk < NTOK) {
          const int i1 = nq / WS7, j1 = nq % WS7;
          const int i2 = mk / WS7, j2 = mk % WS7;
          const float bias = rpb[((i1 - i2 + 6) * 13 + (j1 - j2 + 6)) * NH + head];
          const int r1 = regid(wwh * WS7 + i1) * 3 + regid(www * WS7 + j1);
          const int r2 = regid(wwh * WS7 + i2) * 3 + regid(www * WS7 + j2);
          sval = acc[r] + bias + ((r1 == r2) ? 0.f : -100.f);
        }
        S[nq * NPAD + mk] = sval;
      }
    }
  }
  __syncthreads();

  // softmax rows (fp32) -> P (f16), pad cols zeroed
  for (int rr = lane; rr < NPAD; rr += 32) {
    float mx = -1e30f;
    for (int c = 0; c < NTOK; ++c) mx = fmaxf(mx, S[rr * NPAD + c]);
    float sum = 0.f;
    for (int c = 0; c < NTOK; ++c) sum += __expf(S[rr * NPAD + c] - mx);
    const float inv = 1.f / sum;
    for (int c = 0; c < NPAD; ++c)
      P[rr * NPAD + c] = (c < NTOK) ? (_Float16)(__expf(S[rr * NPAD + c] - mx) * inv)
                                    : (_Float16)0.f;
  }
  __syncthreads();

  // B-fragments of V: b[i] = V[m = ks*32 + lhi*16 + i][d = nt*16+lrow] from vT
  v16h vf[2][2];
#pragma unroll
  for (int ks = 0; ks < 2; ++ks)
#pragma unroll
    for (int nt = 0; nt < 2; ++nt)
      vf[ks][nt] = *(const v16h*)(vbase + (nt * 16 + lrow) * NPAD + ks * 32 + lhi * 16);

  // O = P V  (K = 64 -> two WMMA K-steps); pad query rows get explicit zeros
#pragma unroll
  for (int mt = 0; mt < 4; ++mt) {
    v8f o0 = {}, o1 = {};
#pragma unroll
    for (int ks = 0; ks < 2; ++ks) {
      const _Float16* pp = &P[(mt * 16 + lrow) * NPAD + ks * 32 + lhi * 16];
      v16h pf = cat8(*(const v8h*)pp, *(const v8h*)(pp + 8));
      o0 = WMMA_F16(pf, vf[ks][0], o0);
      o1 = WMMA_F16(pf, vf[ks][1], o1);
    }
#pragma unroll
    for (int r = 0; r < 8; ++r) {
      const int nq = mt * 16 + r + lhi * 8;
      const bool valid = (nq < NTOK);
      _Float16* dst = ao + ((size_t)win * NPAD + nq) * C_DIM + head * HD;
      dst[lrow]      = valid ? (_Float16)o0[r] : (_Float16)0.f;
      dst[16 + lrow] = valid ? (_Float16)o1[r] : (_Float16)0.f;
    }
  }
}

// =====================================================================
// window reverse + residual + LN2: wave per token
// =====================================================================
__global__ __launch_bounds__(256) void res_ln2_kernel(
    const float* __restrict__ x, const float* __restrict__ po,
    const float* __restrict__ g, const float* __restrict__ b,
    float* __restrict__ x1, _Float16* __restrict__ y) {
  const int token = blockIdx.x * 8 + (threadIdx.x >> 5);
  const int lane  = threadIdx.x & 31;
  const int bi = token / (HW * HW);
  const int hw = token % (HW * HW);
  int hp = hw / HW - SS3; if (hp < 0) hp += HW;
  int wp = hw % HW - SS3; if (wp < 0) wp += HW;
  const int win = (hp / WS7) * 8 + (wp / WS7);
  const int n   = (hp % WS7) * WS7 + (wp % WS7);
  const size_t m = ((size_t)bi * NWIN + win) * NPAD + n;

  float v[12], s = 0.f, s2 = 0.f;
#pragma unroll
  for (int i = 0; i < 12; ++i) {
    const int c = lane * 12 + i;
    v[i] = x[(size_t)token * C_DIM + c] + po[m * C_DIM + c];
    s += v[i]; s2 += v[i] * v[i];
  }
#pragma unroll
  for (int o = 16; o > 0; o >>= 1) { s += __shfl_xor(s, o, 32); s2 += __shfl_xor(s2, o, 32); }
  const float mu = s * (1.f / C_DIM);
  const float rs = rsqrtf(s2 * (1.f / C_DIM) - mu * mu + 1e-5f);
#pragma unroll
  for (int i = 0; i < 12; ++i) {
    const int c = lane * 12 + i;
    x1[(size_t)token * C_DIM + c] = v[i];
    y [(size_t)token * C_DIM + c] = (_Float16)((v[i] - mu) * rs * g[c] + b[c]);
  }
}

// =====================================================================
// launch
// =====================================================================
extern "C" void kernel_launch(void* const* d_in, const int* in_sizes, int n_in,
                              void* d_out, int out_size, void* d_ws, size_t ws_size,
                              hipStream_t stream) {
  (void)in_sizes; (void)n_in; (void)out_size; (void)ws_size;
  const float* x      = (const float*)d_in[0];
  const float* qkv_w  = (const float*)d_in[1];
  const float* qkv_b  = (const float*)d_in[2];
  const float* proj_w = (const float*)d_in[3];
  const float* proj_b = (const float*)d_in[4];
  const float* rpb    = (const float*)d_in[5];
  const float* n1g    = (const float*)d_in[6];
  const float* n1b    = (const float*)d_in[7];
  const float* n2g    = (const float*)d_in[8];
  const float* n2b    = (const float*)d_in[9];
  const float* w1     = (const float*)d_in[10];
  const float* b1     = (const float*)d_in[11];
  const float* w2     = (const float*)d_in[12];
  const float* b2     = (const float*)d_in[13];
  float* out = (float*)d_out;

  char* ws = (char*)d_ws;
  // f16 weight copies (~3.5 MB)
  _Float16* qkv_wh  = (_Float16*)(ws + 0);              // 1152*384
  _Float16* proj_wh = (_Float16*)(ws + 884736);         // 384*384
  _Float16* w1h     = (_Float16*)(ws + 1179648);        // 1536*384
  _Float16* w2h     = (_Float16*)(ws + 2359296);        // 384*1536
  // buf1: xw -> ao -> y (100.7 MB)
  char* buf1 = ws + 3538944;
  // buf2: {q,k,vT} -> po(f32) -> h(f16)  (308.3 MB)
  char* buf2 = buf1 + 100663296;
  // buf3: x1 f32 (154.1 MB)
  char* buf3 = buf2 + 308281344;

  _Float16* xw  = (_Float16*)buf1;
  _Float16* ao  = (_Float16*)buf1;
  _Float16* y   = (_Float16*)buf1;
  _Float16* qB  = (_Float16*)buf2;
  _Float16* kB  = (_Float16*)(buf2 + 100663296);
  _Float16* vT  = (_Float16*)(buf2 + 201326592);
  float*    po  = (float*)buf2;
  _Float16* hB  = (_Float16*)buf2;
  float*    x1  = (float*)buf3;

  // 1) weights -> f16 (keep [N,K] row-major = native B-fragment layout)
  cvt_f32_f16<<<(1152 * 384 + 255) / 256, 256, 0, stream>>>(qkv_w, qkv_wh, 1152 * 384);
  cvt_f32_f16<<<(384 * 384 + 255) / 256, 256, 0, stream>>>(proj_w, proj_wh, 384 * 384);
  cvt_f32_f16<<<(1536 * 384 + 255) / 256, 256, 0, stream>>>(w1, w1h, 1536 * 384);
  cvt_f32_f16<<<(384 * 1536 + 255) / 256, 256, 0, stream>>>(w2, w2h, 384 * 1536);

  // 2) zero xw (pad rows must be 0), then LN1 + shift + window partition
  fill_zero4<<<24576, 256, 0, stream>>>((uint4*)buf1, 100663296 / 16);
  ln1_shift_window_kernel<<<NTOKENS / 8, 256, 0, stream>>>(x, n1g, n1b, xw);

  // 3) QKV GEMM: M=131072, N=1152, K=384
  gemm_qkv_kernel<<<dim3(1152 / 128, MPAD / 128), 256, 0, stream>>>(
      xw, qkv_wh, qkv_b, qB, kB, vT);

  // 4) attention per (window, head); writes every ao element (pad rows = 0)
  attn_kernel<<<NWINTOT * NH, 32, 0, stream>>>(qB, kB, vT, rpb, ao);

  // 5) proj GEMM: M=131072, N=384, K=384 -> fp32
  gemm_proj_kernel<<<dim3(384 / 128, MPAD / 128), 256, 0, stream>>>(ao, proj_wh, proj_b, po);

  // 6) window reverse + residual + LN2
  res_ln2_kernel<<<NTOKENS / 8, 256, 0, stream>>>(x, po, n2g, n2b, x1, y);

  // 7) fc1 + GELU: M=100352, N=1536, K=384
  gemm_fc1_gelu_kernel<<<dim3(1536 / 128, NTOKENS / 128), 256, 0, stream>>>(y, w1h, b1, hB);

  // 8) fc2 + residual: M=100352, N=384, K=1536 -> d_out
  gemm_fc2_res_kernel<<<dim3(384 / 128, NTOKENS / 128), 256, 0, stream>>>(hB, w2h, b2, x1, out);
}
